// GNNModel_25305947308764
// MI455X (gfx1250) — compile-verified, matrix operand order checked by reading, not run on hardware
//
#include <hip/hip_runtime.h>

typedef __attribute__((ext_vector_type(16))) _Float16 v16h;
typedef __attribute__((ext_vector_type(8)))  float    v8f;

#define HDIM 32

// ---------------- utility kernels ----------------

__global__ void zero_f32(float* __restrict__ p, int n) {
  int i = blockIdx.x * blockDim.x + threadIdx.x;
  int stride = gridDim.x * blockDim.x;
  for (; i < n; i += stride) p[i] = 0.0f;
}

__global__ void deg_kernel(const int* __restrict__ dst, float* __restrict__ deg, int E) {
  int e = blockIdx.x * blockDim.x + threadIdx.x;
  if (e < E) unsafeAtomicAdd(&deg[dst[e]], 1.0f);
}

__global__ void dinv_kernel(const float* __restrict__ deg, float* __restrict__ dinv, int N) {
  int n = blockIdx.x * blockDim.x + threadIdx.x;
  if (n < N) dinv[n] = rsqrtf(deg[n] + 1.0f);
}

// Pre-swizzle a [F,32] f32 weight into per-lane WMMA B fragments (f16).
// Output layout: P[lane*32 + 0..15]  = b0 fragment (cols 0..15 tile)
//                P[lane*32 + 16..31] = b1 fragment (cols 16..31 tile)
__global__ void pack_wfrag(const float* __restrict__ W, _Float16* __restrict__ P, int F) {
  int t = blockIdx.x * blockDim.x + threadIdx.x;
  if (t >= 32 * 32) return;
  int lane = t >> 5;
  int idx  = t & 31;
  int m  = lane & 15;
  int hi = lane >> 4;
  int kb = hi * 8;
  int frag = idx >> 4;                 // 0 -> b0, 1 -> b1
  int j    = idx & 15;                 // slot within v16h
  int k    = (j < 8) ? (kb + j) : (16 + kb + (j - 8));
  int col  = m + 16 * frag;
  P[t] = (k < F) ? (_Float16)W[k * HDIM + col] : (_Float16)0.0f;
}

// ---------------- WMMA feature transform: out[N,32] = in[N,F] @ W[F,32] ----------------
// One wave handles 16 rows; F is compile-time (16 or 32).

template <int F>
__global__ __launch_bounds__(256) void wmma_xform(const float* __restrict__ in,
                                                  const _Float16* __restrict__ Wp,
                                                  float* __restrict__ out, int N) {
  int lane = threadIdx.x & 31;
  int wave = threadIdx.x >> 5;
  int base = (blockIdx.x * 8 + wave) * 16;
  int m  = lane & 15;
  int hi = lane >> 4;
  int kb = hi * 8;

  int row  = base + m;
  int rowc = (row < N) ? row : 0;
  const float4* rp4 = (const float4*)(in + (size_t)rowc * (size_t)F);

  float lo[8], hv[8];
  *(float4*)&lo[0] = rp4[(kb >> 2) + 0];
  *(float4*)&lo[4] = rp4[(kb >> 2) + 1];
  if (F > 16) {
    *(float4*)&hv[0] = rp4[((16 + kb) >> 2) + 0];
    *(float4*)&hv[4] = rp4[((16 + kb) >> 2) + 1];
  } else {
#pragma unroll
    for (int i = 0; i < 8; ++i) hv[i] = 0.0f;
  }

  v16h a;
#pragma unroll
  for (int i = 0; i < 8; ++i) {
    a[i]     = (_Float16)lo[i];
    a[8 + i] = (_Float16)hv[i];
  }

  const v16h* wp = (const v16h*)Wp + lane * 2;
  v16h b0 = wp[0];
  v16h b1 = wp[1];

  v8f c0 = {}, c1 = {};
  c0 = __builtin_amdgcn_wmma_f32_16x16x32_f16(false, a, false, b0, (short)0, c0, false, false);
  c1 = __builtin_amdgcn_wmma_f32_16x16x32_f16(false, a, false, b1, (short)0, c1, false, false);

  if (base + 16 <= N) {
    float* op = out + (size_t)(base + 8 * hi) * HDIM + m;
#pragma unroll
    for (int r = 0; r < 8; ++r) {
      op[r * HDIM]      = c0[r];
      op[r * HDIM + 16] = c1[r];
    }
  } else {
#pragma unroll
    for (int r = 0; r < 8; ++r) {
      int orow = base + r + 8 * hi;
      if (orow < N) {
        out[(size_t)orow * HDIM + m]      = c0[r];
        out[(size_t)orow * HDIM + 16 + m] = c1[r];
      }
    }
  }
}

// ---------------- edge scatter: agg[dst] += xw[src] * dinv[src]*dinv[dst] ----------------

__global__ void scatter_kernel(const int* __restrict__ src, const int* __restrict__ dst,
                               const float* __restrict__ xw, const float* __restrict__ dinv,
                               float* __restrict__ agg, int E) {
  long long t = (long long)blockIdx.x * blockDim.x + threadIdx.x;
  if (t >= (long long)E * HDIM) return;
  int e = (int)(t >> 5);
  int j = (int)(t & 31);
  int s = src[e], d = dst[e];
  float coef = dinv[s] * dinv[d];
  unsafeAtomicAdd(&agg[(size_t)d * HDIM + j], xw[(size_t)s * HDIM + j] * coef);
}

// ---------------- finish: out = agg + xw*dinv^2 + b  (optional relu) ----------------

__global__ void finish_layer(const float* __restrict__ xw, const float* __restrict__ agg,
                             const float* __restrict__ dinv, const float* __restrict__ b,
                             float* __restrict__ out, int N, int do_relu) {
  long long t = (long long)blockIdx.x * blockDim.x + threadIdx.x;
  if (t >= (long long)N * HDIM) return;
  int n = (int)(t >> 5);
  int j = (int)(t & 31);
  float di = dinv[n];
  float v = agg[t] + xw[t] * di * di + b[j];
  if (do_relu) v = fmaxf(v, 0.0f);
  out[t] = v;
}

// ---------------- global mean pool + regression head ----------------

__global__ void pool_kernel(const float* __restrict__ h, const int* __restrict__ batch,
                            float* __restrict__ pooled, float* __restrict__ cnt, int N) {
  long long t = (long long)blockIdx.x * blockDim.x + threadIdx.x;
  if (t >= (long long)N * HDIM) return;
  int n = (int)(t >> 5);
  int j = (int)(t & 31);
  int g = batch[n];
  unsafeAtomicAdd(&pooled[(size_t)g * HDIM + j], h[t]);
  if (j == 0) unsafeAtomicAdd(&cnt[g], 1.0f);
}

__global__ void reg_kernel(const float* __restrict__ pooled, const float* __restrict__ cnt,
                           const float* __restrict__ Wr, const float* __restrict__ br,
                           float* __restrict__ out, int G) {
  int g = blockIdx.x * blockDim.x + threadIdx.x;
  if (g >= G) return;
  float inv = 1.0f / fmaxf(cnt[g], 1.0f);
  float acc = 0.0f;
#pragma unroll
  for (int j = 0; j < HDIM; ++j) acc += pooled[(size_t)g * HDIM + j] * inv * Wr[j];
  out[g] = acc + br[0];
}

// ---------------- bilinear edge head: out_e = h[src_e]^T Wbil h[dst_e] + b ----------------
// One wave = 16 edges. T = H1 @ Wbil via 2 WMMAs, then rowwise f32 dot with H2.

__global__ __launch_bounds__(256) void bilinear_kernel(const float* __restrict__ h,
                                                       const int* __restrict__ srcA,
                                                       const int* __restrict__ dstA,
                                                       const _Float16* __restrict__ Wbp,
                                                       const float* __restrict__ bbil,
                                                       float* __restrict__ out, int EA) {
  __shared__ float T[8][16][HDIM];
  __shared__ float P[8][16];

  int lane = threadIdx.x & 31;
  int wave = threadIdx.x >> 5;
  int base = (blockIdx.x * 8 + wave) * 16;
  int m  = lane & 15;
  int hi = lane >> 4;
  int kb = hi * 8;

  int e  = base + m;
  int ec = (e < EA) ? e : 0;
  int s  = srcA[ec];
  const float4* hp4 = (const float4*)(h + (size_t)s * HDIM);

  float lo[8], hv[8];
  *(float4*)&lo[0] = hp4[(kb >> 2) + 0];
  *(float4*)&lo[4] = hp4[(kb >> 2) + 1];
  *(float4*)&hv[0] = hp4[((16 + kb) >> 2) + 0];
  *(float4*)&hv[4] = hp4[((16 + kb) >> 2) + 1];

  v16h a;
#pragma unroll
  for (int i = 0; i < 8; ++i) {
    a[i]     = (_Float16)lo[i];
    a[8 + i] = (_Float16)hv[i];
  }

  const v16h* wp = (const v16h*)Wbp + lane * 2;
  v16h b0 = wp[0];
  v16h b1 = wp[1];

  v8f c0 = {}, c1 = {};
  c0 = __builtin_amdgcn_wmma_f32_16x16x32_f16(false, a, false, b0, (short)0, c0, false, false);
  c1 = __builtin_amdgcn_wmma_f32_16x16x32_f16(false, a, false, b1, (short)0, c1, false, false);

#pragma unroll
  for (int r = 0; r < 8; ++r) {
    int M = r + 8 * hi;
    T[wave][M][m]      = c0[r];
    T[wave][M][16 + m] = c1[r];
  }
  __syncthreads();

  int d = dstA[ec];
  const float4* h24 = (const float4*)(h + (size_t)d * HDIM) + hi * 4;
  float t2[16];
  *(float4*)&t2[0]  = h24[0];
  *(float4*)&t2[4]  = h24[1];
  *(float4*)&t2[8]  = h24[2];
  *(float4*)&t2[12] = h24[3];

  float acc = 0.0f;
#pragma unroll
  for (int j = 0; j < 16; ++j) acc += T[wave][m][hi * 16 + j] * t2[j];

  if (hi) P[wave][m] = acc;
  __syncthreads();
  if (!hi && e < EA) out[e] = acc + P[wave][m] + bbil[0];
}

// ---------------- host orchestration ----------------

static inline int gdiv(long long a, long long b) { return (int)((a + b - 1) / b); }

extern "C" void kernel_launch(void* const* d_in, const int* in_sizes, int n_in,
                              void* d_out, int out_size, void* d_ws, size_t ws_size,
                              hipStream_t stream) {
  const float* x     = (const float*)d_in[0];
  const int*   ei    = (const int*)d_in[1];
  const int*   eia   = (const int*)d_in[2];
  const int*   batch = (const int*)d_in[3];
  const float* W1    = (const float*)d_in[4];
  const float* b1    = (const float*)d_in[5];
  const float* W2    = (const float*)d_in[6];
  const float* b2    = (const float*)d_in[7];
  const float* Wr    = (const float*)d_in[8];
  const float* br    = (const float*)d_in[9];
  const float* Wbil  = (const float*)d_in[10];
  const float* bbil  = (const float*)d_in[11];

  const int N  = in_sizes[3];
  const int F  = in_sizes[0] / N;      // 16
  const int E  = in_sizes[1] / 2;
  const int EA = in_sizes[2] / 2;
  const int G  = out_size - EA;        // 512

  const int* src  = ei;
  const int* dst  = ei + E;
  const int* srcA = eia;
  const int* dstA = eia + EA;

  float* ws     = (float*)d_ws;
  float* deg    = ws;                              // N
  float* dinv   = deg + N;                         // N
  float* xw     = dinv + N;                        // N*32
  float* agg    = xw + (size_t)N * HDIM;           // N*32
  float* hbuf   = agg + (size_t)N * HDIM;          // N*32
  float* pooled = hbuf + (size_t)N * HDIM;         // G*32
  float* cnt    = pooled + (size_t)G * HDIM;       // G
  // align following f16 pack buffers to 64B
  size_t off = (size_t)(cnt + G - ws);
  off = (off + 15) & ~(size_t)15;
  _Float16* W1p = (_Float16*)(ws + off);           // 1024 halves (2KB)
  _Float16* W2p = W1p + 32 * 32;                   // 1024 halves
  _Float16* Wbp = W2p + 32 * 32;                   // 1024 halves

  float* out_reg  = (float*)d_out;
  float* out_edge = (float*)d_out + G;

  const int B = 256;

  // pre-swizzle weights into per-lane WMMA fragments
  pack_wfrag<<<gdiv(1024, B), B, 0, stream>>>(W1, W1p, F);
  pack_wfrag<<<gdiv(1024, B), B, 0, stream>>>(W2, W2p, HDIM);
  pack_wfrag<<<gdiv(1024, B), B, 0, stream>>>(Wbil, Wbp, HDIM);

  // degree / dinv
  zero_f32<<<gdiv(N, B), B, 0, stream>>>(deg, N);
  deg_kernel<<<gdiv(E, B), B, 0, stream>>>(dst, deg, E);
  dinv_kernel<<<gdiv(N, B), B, 0, stream>>>(deg, dinv, N);

  const int xform_blocks = gdiv(gdiv(N, 16), 8);

  // ---- layer 1 ----
  wmma_xform<16><<<xform_blocks, B, 0, stream>>>(x, W1p, xw, N);
  zero_f32<<<gdiv((long long)N * HDIM, B), B, 0, stream>>>(agg, N * HDIM);
  scatter_kernel<<<gdiv((long long)E * HDIM, B), B, 0, stream>>>(src, dst, xw, dinv, agg, E);
  finish_layer<<<gdiv((long long)N * HDIM, B), B, 0, stream>>>(xw, agg, dinv, b1, hbuf, N, 1);

  // ---- layer 2 ----
  wmma_xform<32><<<xform_blocks, B, 0, stream>>>(hbuf, W2p, xw, N);
  zero_f32<<<gdiv((long long)N * HDIM, B), B, 0, stream>>>(agg, N * HDIM);
  scatter_kernel<<<gdiv((long long)E * HDIM, B), B, 0, stream>>>(src, dst, xw, dinv, agg, E);
  finish_layer<<<gdiv((long long)N * HDIM, B), B, 0, stream>>>(xw, agg, dinv, b2, hbuf, N, 0);

  // ---- mean pool + regression head ----
  zero_f32<<<gdiv(G * (HDIM + 1), B), B, 0, stream>>>(pooled, G * (HDIM + 1)); // pooled + cnt (contiguous)
  pool_kernel<<<gdiv((long long)N * HDIM, B), B, 0, stream>>>(hbuf, batch, pooled, cnt, N);
  reg_kernel<<<gdiv(G, B), B, 0, stream>>>(pooled, cnt, Wr, br, out_reg, G);

  // ---- bilinear edge head ----
  const int bil_blocks = gdiv(gdiv(EA, 16), 8);
  bilinear_kernel<<<bil_blocks, B, 0, stream>>>(hbuf, srcA, dstA, Wbp, bbil, out_edge, EA);
}
